// SparseConvCausalAttention_64183991272015
// MI455X (gfx1250) — matrix operand order repaired and checked by reading, not compiled, not run on hardware
//
#include <hip/hip_runtime.h>

// ---------------------------------------------------------------------------
// Types for CDNA5 WMMA (wave32): bf16 16x16x32, f32 accumulate.
// ---------------------------------------------------------------------------
typedef __attribute__((ext_vector_type(16))) __bf16 bf16x16;
typedef __attribute__((ext_vector_type(8)))  __bf16 bf16x8;
typedef __attribute__((ext_vector_type(8)))  float  floatx8;

#define B_     4
#define N_     1279
#define DIM_   512
#define HEADS_ 8
#define DH_    64
#define L_     1280
#define TEXT_  256
#define INNER_ 512
#define QKV3_  1536
#define BH_    32
#define NEG_   (-3.0e38f)

static __device__ __forceinline__ floatx8 wmma_bf16(bf16x16 a, bf16x16 b, floatx8 c) {
  return __builtin_amdgcn_wmma_f32_16x16x32_bf16(false, a, false, b, (short)0, c,
                                                 false, false);
}

// A-fragment (16x32 bf16): lane l holds row (l&15); elements 0..15 are the
// 16 consecutive K at +((l>>4)*16). Two 16-byte
// loads. Caller passes row pointer already offset by (l>>4)*16.
static __device__ __forceinline__ bf16x16 load_a_frag(const __bf16* p) {
  union { bf16x16 v; bf16x8 h[2]; } u;
  u.h[0] = *(const bf16x8*)p;
  u.h[1] = *(const bf16x8*)(p + 8);
  return u.v;
}

// B-fragment (32x16 bf16): lane l holds column (l&15); elements 0..15 are the
// 16 consecutive K rows at +((l>>4)*16). One 32-byte load from a K-contiguous
// ("transposed") layout.
static __device__ __forceinline__ bf16x16 load_b_frag(const __bf16* p) {
  return *(const bf16x16*)p;
}

// ---------------------------------------------------------------------------
// Kernel 1: f32 -> bf16 conversions. x padded to L rows; W_qkv / W_out stored
// transposed (output-feature-major) so WMMA B fragments are contiguous.
// ---------------------------------------------------------------------------
__global__ void k_convert(const float* __restrict__ x,
                          const float* __restrict__ wqkv,
                          const float* __restrict__ wout,
                          __bf16* __restrict__ xb,
                          __bf16* __restrict__ wqkvt,
                          __bf16* __restrict__ woutt) {
  const int T1 = B_ * L_ * DIM_;
  const int T2 = QKV3_ * DIM_;
  const int T3 = INNER_ * DIM_;
  for (int i = blockIdx.x * blockDim.x + threadIdx.x; i < T1 + T2 + T3;
       i += gridDim.x * blockDim.x) {
    if (i < T1) {
      int row = i / DIM_, d = i % DIM_;
      int b = row / L_, s = row % L_;
      float v = (s < N_) ? x[((size_t)b * N_ + s) * DIM_ + d] : 0.0f;
      xb[i] = (__bf16)v;
    } else if (i < T1 + T2) {
      int j = i - T1;
      int n = j / DIM_, k = j % DIM_;
      wqkvt[j] = (__bf16)wqkv[(size_t)k * QKV3_ + n];
    } else {
      int j = i - T1 - T2;
      int n = j / DIM_, k = j % DIM_;
      woutt[j] = (__bf16)wout[(size_t)k * INNER_ + n];
    }
  }
}

// ---------------------------------------------------------------------------
// Kernel 2: QKV projection GEMM (5120x512 @ 512x1536) via WMMA.
// Each wave owns a 16(M) x 64(N) tile; K in steps of 32 => 64 WMMAs/wave.
// Epilogue scatters head-major bf16 q (pre-scaled by DH^-0.5), k, and
// transposed v (vt[bh][d][seq]) for contiguous P*V B-fragments later.
// ---------------------------------------------------------------------------
__global__ __launch_bounds__(128) void k_qkv(const __bf16* __restrict__ xb,
                                             const __bf16* __restrict__ wt,
                                             __bf16* __restrict__ qh,
                                             __bf16* __restrict__ kh,
                                             __bf16* __restrict__ vt) {
  int wave = threadIdx.x >> 5, lane = threadIdx.x & 31;
  int tile = blockIdx.x * 4 + wave;          // 320 * 24 = 7680 tiles
  int mt = tile % 320, ns = tile / 320;
  int m0 = mt * 16, n0 = ns * 64;
  int lr = lane & 15, lh = lane >> 4;

  floatx8 acc0 = {}, acc1 = {}, acc2 = {}, acc3 = {};
  const __bf16* arow = xb + (size_t)(m0 + lr) * DIM_ + lh * 16;
  const __bf16* b0 = wt + (size_t)(n0 +  0 + lr) * DIM_ + lh * 16;
  const __bf16* b1 = wt + (size_t)(n0 + 16 + lr) * DIM_ + lh * 16;
  const __bf16* b2 = wt + (size_t)(n0 + 32 + lr) * DIM_ + lh * 16;
  const __bf16* b3 = wt + (size_t)(n0 + 48 + lr) * DIM_ + lh * 16;

  for (int k0 = 0; k0 < DIM_; k0 += 32) {
    __builtin_prefetch(arow + k0 + 64, 0, 1);   // global_prefetch_b8
    bf16x16 a = load_a_frag(arow + k0);
    acc0 = wmma_bf16(a, load_b_frag(b0 + k0), acc0);
    acc1 = wmma_bf16(a, load_b_frag(b1 + k0), acc1);
    acc2 = wmma_bf16(a, load_b_frag(b2 + k0), acc2);
    acc3 = wmma_bf16(a, load_b_frag(b3 + k0), acc3);
  }

  floatx8 accs[4] = {acc0, acc1, acc2, acc3};
#pragma unroll
  for (int t = 0; t < 4; ++t) {
#pragma unroll
    for (int r = 0; r < 8; ++r) {
      int m = m0 + r + 8 * lh;
      int n = n0 + t * 16 + lr;
      float v = accs[t][r];
      int s = m % L_, b = m / L_;
      int which = n >> 9, h = (n >> 6) & 7, d = n & 63;
      int bh = b * HEADS_ + h;
      if (which == 0)
        qh[((size_t)bh * L_ + s) * DH_ + d] = (__bf16)(v * 0.125f);  // * DH^-0.5
      else if (which == 1)
        kh[((size_t)bh * L_ + s) * DH_ + d] = (__bf16)v;
      else
        vt[((size_t)bh * DH_ + d) * L_ + s] = (__bf16)v;
    }
  }
}

// ---------------------------------------------------------------------------
// Kernel 3: causal text attention, flash-style. One wave per (bh, 16-query
// tile). 32 keys per iteration: 4 QK^T WMMAs, online softmax in the C-layout
// (row = r + 8*(lane/16), col = lane&15; row reduce = shfl_xor over 16-lane
// halves), P re-shaped to the A layout through per-wave LDS (in-wave DS ops
// are in-order -> no barrier), then 4 P*V WMMAs.
// ---------------------------------------------------------------------------
__global__ __launch_bounds__(128) void k_attn_text(const __bf16* __restrict__ qh,
                                                   const __bf16* __restrict__ kh,
                                                   const __bf16* __restrict__ vt,
                                                   __bf16* __restrict__ ao) {
  __shared__ __attribute__((aligned(64))) __bf16 Pt[4][16][32];
  int wave = threadIdx.x >> 5, lane = threadIdx.x & 31;
  int tile = blockIdx.x * 4 + wave;          // 32 * 16 = 512 tiles
  int bh = tile >> 4, qt = tile & 15;
  int lr = lane & 15, lh = lane >> 4;

  const __bf16* qb = qh + (size_t)bh * L_ * DH_;
  const __bf16* kb = kh + (size_t)bh * L_ * DH_;
  const __bf16* vb = vt + (size_t)bh * DH_ * L_;

  bf16x16 qf0 = load_a_frag(qb + (size_t)(qt * 16 + lr) * DH_ + lh * 16);
  bf16x16 qf1 = load_a_frag(qb + (size_t)(qt * 16 + lr) * DH_ + 32 + lh * 16);

  float rmax[8], rsum[8];
  floatx8 o0 = {}, o1 = {}, o2 = {}, o3 = {};
#pragma unroll
  for (int r = 0; r < 8; ++r) { rmax[r] = NEG_; rsum[r] = 0.0f; }

  int nblk = (qt * 16 + 15) / 32 + 1;        // causal: only key blocks <= diag
  for (int jb = 0; jb < nblk; ++jb) {
    int kk0 = jb * 32;
    floatx8 sa = {}, sb = {};
    {
      const __bf16* kr = kb + (size_t)(kk0 + lr) * DH_ + lh * 16;
      sa = wmma_bf16(qf0, load_b_frag(kr), sa);
      sa = wmma_bf16(qf1, load_b_frag(kr + 32), sa);
      kr += 16 * DH_;
      sb = wmma_bf16(qf0, load_b_frag(kr), sb);
      sb = wmma_bf16(qf1, load_b_frag(kr + 32), sb);
    }
#pragma unroll
    for (int r = 0; r < 8; ++r) {
      int q = qt * 16 + r + 8 * lh;
      float s0 = (kk0 + lr > q) ? NEG_ : sa[r];
      float s1 = (kk0 + 16 + lr > q) ? NEG_ : sb[r];
      float mx = fmaxf(s0, s1);
      mx = fmaxf(mx, __shfl_xor(mx, 1));
      mx = fmaxf(mx, __shfl_xor(mx, 2));
      mx = fmaxf(mx, __shfl_xor(mx, 4));
      mx = fmaxf(mx, __shfl_xor(mx, 8));
      float mnew = fmaxf(rmax[r], mx);
      float corr = __expf(rmax[r] - mnew);
      rmax[r] = mnew;
      float p0 = __expf(s0 - mnew), p1 = __expf(s1 - mnew);
      float ps = p0 + p1;
      ps += __shfl_xor(ps, 1);
      ps += __shfl_xor(ps, 2);
      ps += __shfl_xor(ps, 4);
      ps += __shfl_xor(ps, 8);
      rsum[r] = rsum[r] * corr + ps;
      o0[r] *= corr; o1[r] *= corr; o2[r] *= corr; o3[r] *= corr;
      Pt[wave][r + 8 * lh][lr]      = (__bf16)p0;
      Pt[wave][r + 8 * lh][16 + lr] = (__bf16)p1;
    }
    bf16x16 pf = load_a_frag(&Pt[wave][lr][lh * 16]);
    o0 = wmma_bf16(pf, load_b_frag(vb + (size_t)(lr)      * L_ + kk0 + lh * 16), o0);
    o1 = wmma_bf16(pf, load_b_frag(vb + (size_t)(16 + lr) * L_ + kk0 + lh * 16), o1);
    o2 = wmma_bf16(pf, load_b_frag(vb + (size_t)(32 + lr) * L_ + kk0 + lh * 16), o2);
    o3 = wmma_bf16(pf, load_b_frag(vb + (size_t)(48 + lr) * L_ + kk0 + lh * 16), o3);
  }

  int b = bh >> 3, h = bh & 7;
#pragma unroll
  for (int r = 0; r < 8; ++r) {
    float inv = 1.0f / rsum[r];
    int s = qt * 16 + r + 8 * lh;
    size_t rowoff = ((size_t)b * L_ + s) * INNER_ + h * DH_;
    ao[rowoff + lr]      = (__bf16)(o0[r] * inv);
    ao[rowoff + 16 + lr] = (__bf16)(o1[r] * inv);
    ao[rowoff + 32 + lr] = (__bf16)(o2[r] * inv);
    ao[rowoff + 48 + lr] = (__bf16)(o3[r] * inv);
  }
}

// ---------------------------------------------------------------------------
// Kernel 4: image attention = causal 5x5 local window (irregular gather,
// scalar, folded into the online softmax via an LDS funnel) + image->text
// attention (8 x 32-key WMMA flash blocks with the text mask). One wave per
// (bh, 16-query tile).
// ---------------------------------------------------------------------------
__global__ __launch_bounds__(128) void k_attn_img(const __bf16* __restrict__ qh,
                                                  const __bf16* __restrict__ kh,
                                                  const __bf16* __restrict__ vt,
                                                  const unsigned char* __restrict__ mask,
                                                  __bf16* __restrict__ ao) {
  __shared__ __attribute__((aligned(64))) float  Pl[4][16][32];
  __shared__ __attribute__((aligned(64))) __bf16 Pt[4][16][32];
  int wave = threadIdx.x >> 5, lane = threadIdx.x & 31;
  int tile = blockIdx.x * 4 + wave;          // 32 * 64 = 2048 tiles
  int bh = tile >> 6, it = tile & 63;
  int iq0 = it * 16;
  int lr = lane & 15, lh = lane >> 4;
  int b = bh >> 3, h = bh & 7;

  const __bf16* qb = qh + (size_t)bh * L_ * DH_;
  const __bf16* kb = kh + (size_t)bh * L_ * DH_;
  const __bf16* vb = vt + (size_t)bh * DH_ * L_;

  // ---- 1) local 5x5 window logits (lane (lr) owns query lr, half-waves
  //         split the 25 taps). q is pre-scaled, zero-pad + causal => -inf.
  {
    int mq = lr;
    int qi = iq0 + mq, qy = qi >> 5, qx = qi & 31;
    const __bf16* qr = qb + (size_t)(TEXT_ + qi) * DH_;
    for (int j = lh; j < 25; j += 2) {
      int dy = j / 5 - 2, dx = j % 5 - 2;
      int ny = qy + dy, nx = qx + dx;
      float logit = NEG_;
      if (ny >= 0 && ny < 32 && nx >= 0 && nx < 32) {
        int p = ny * 32 + nx;
        if (qi >= p) {                        // causal within the image
          const __bf16* krow = kb + (size_t)(TEXT_ + p) * DH_;
          float dot = 0.0f;
#pragma unroll 16
          for (int d = 0; d < DH_; ++d) dot += (float)qr[d] * (float)krow[d];
          logit = dot;
        }
      }
      Pl[wave][mq][j] = logit;
    }
    if (lh == 0) {
#pragma unroll
      for (int j = 25; j < 32; ++j) Pl[wave][mq][j] = NEG_;
    }
  }

  // ---- 2) fold local logits into online-softmax state (read in C-layout)
  float rmax[8], rsum[8];
  floatx8 o0 = {}, o1 = {}, o2 = {}, o3 = {};
#pragma unroll
  for (int r = 0; r < 8; ++r) {
    float s0 = Pl[wave][r + 8 * lh][lr];
    float s1 = Pl[wave][r + 8 * lh][16 + lr];
    float mx = fmaxf(s0, s1);
    mx = fmaxf(mx, __shfl_xor(mx, 1));
    mx = fmaxf(mx, __shfl_xor(mx, 2));
    mx = fmaxf(mx, __shfl_xor(mx, 4));
    mx = fmaxf(mx, __shfl_xor(mx, 8));
    rmax[r] = mx;                             // self tap always valid => finite
    float p0 = __expf(s0 - mx), p1 = __expf(s1 - mx);
    float ps = p0 + p1;
    ps += __shfl_xor(ps, 1);
    ps += __shfl_xor(ps, 2);
    ps += __shfl_xor(ps, 4);
    ps += __shfl_xor(ps, 8);
    rsum[r] = ps;
    Pl[wave][r + 8 * lh][lr]      = p0;       // overwrite logits with probs
    Pl[wave][r + 8 * lh][16 + lr] = p1;
  }

  // ---- 3) gathered P*V for the local window (scalar: V rows differ per query)
  for (int j = 0; j < 25; ++j) {
    int dy = j / 5 - 2, dx = j % 5 - 2;
#pragma unroll
    for (int r = 0; r < 8; ++r) {
      int mq = r + 8 * lh;
      int qi = iq0 + mq;
      int ny = (qi >> 5) + dy, nx = (qi & 31) + dx;
      if (ny < 0 || ny >= 32 || nx < 0 || nx >= 32) continue;
      int p = ny * 32 + nx;
      float pr = Pl[wave][mq][j];
      if (pr > 0.0f) {
        size_t voff = (size_t)TEXT_ + p;
        o0[r] += pr * (float)vb[(size_t)(lr)      * L_ + voff];
        o1[r] += pr * (float)vb[(size_t)(16 + lr) * L_ + voff];
        o2[r] += pr * (float)vb[(size_t)(32 + lr) * L_ + voff];
        o3[r] += pr * (float)vb[(size_t)(48 + lr) * L_ + voff];
      }
    }
  }

  // ---- 4) image -> text attention: 8 WMMA flash blocks over 256 text keys
  bf16x16 qf0 = load_a_frag(qb + (size_t)(TEXT_ + iq0 + lr) * DH_ + lh * 16);
  bf16x16 qf1 = load_a_frag(qb + (size_t)(TEXT_ + iq0 + lr) * DH_ + 32 + lh * 16);
  for (int jb = 0; jb < 8; ++jb) {
    int kk0 = jb * 32;
    floatx8 sa = {}, sb = {};
    const __bf16* kr = kb + (size_t)(kk0 + lr) * DH_ + lh * 16;
    sa = wmma_bf16(qf0, load_b_frag(kr), sa);
    sa = wmma_bf16(qf1, load_b_frag(kr + 32), sa);
    kr += 16 * DH_;
    sb = wmma_bf16(qf0, load_b_frag(kr), sb);
    sb = wmma_bf16(qf1, load_b_frag(kr + 32), sb);
    bool m0v = mask[b * TEXT_ + kk0 + lr] != 0;
    bool m1v = mask[b * TEXT_ + kk0 + 16 + lr] != 0;
#pragma unroll
    for (int r = 0; r < 8; ++r) {
      float s0 = m0v ? sa[r] : NEG_;
      float s1 = m1v ? sb[r] : NEG_;
      float mx = fmaxf(s0, s1);
      mx = fmaxf(mx, __shfl_xor(mx, 1));
      mx = fmaxf(mx, __shfl_xor(mx, 2));
      mx = fmaxf(mx, __shfl_xor(mx, 4));
      mx = fmaxf(mx, __shfl_xor(mx, 8));
      float mnew = fmaxf(rmax[r], mx);
      float corr = __expf(rmax[r] - mnew);
      rmax[r] = mnew;
      float p0 = __expf(s0 - mnew), p1 = __expf(s1 - mnew);
      float ps = p0 + p1;
      ps += __shfl_xor(ps, 1);
      ps += __shfl_xor(ps, 2);
      ps += __shfl_xor(ps, 4);
      ps += __shfl_xor(ps, 8);
      rsum[r] = rsum[r] * corr + ps;
      o0[r] *= corr; o1[r] *= corr; o2[r] *= corr; o3[r] *= corr;
      Pt[wave][r + 8 * lh][lr]      = (__bf16)p0;
      Pt[wave][r + 8 * lh][16 + lr] = (__bf16)p1;
    }
    bf16x16 pf = load_a_frag(&Pt[wave][lr][lh * 16]);
    o0 = wmma_bf16(pf, load_b_frag(vb + (size_t)(lr)      * L_ + kk0 + lh * 16), o0);
    o1 = wmma_bf16(pf, load_b_frag(vb + (size_t)(16 + lr) * L_ + kk0 + lh * 16), o1);
    o2 = wmma_bf16(pf, load_b_frag(vb + (size_t)(32 + lr) * L_ + kk0 + lh * 16), o2);
    o3 = wmma_bf16(pf, load_b_frag(vb + (size_t)(48 + lr) * L_ + kk0 + lh * 16), o3);
  }

#pragma unroll
  for (int r = 0; r < 8; ++r) {
    float inv = 1.0f / rsum[r];
    int s = TEXT_ + iq0 + r + 8 * lh;
    size_t rowoff = ((size_t)b * L_ + s) * INNER_ + h * DH_;
    ao[rowoff + lr]      = (__bf16)(o0[r] * inv);
    ao[rowoff + 16 + lr] = (__bf16)(o1[r] * inv);
    ao[rowoff + 32 + lr] = (__bf16)(o2[r] * inv);
    ao[rowoff + 48 + lr] = (__bf16)(o3[r] * inv);
  }
}

// ---------------------------------------------------------------------------
// Kernel 5: output projection (5120x512 @ 512x512) + bias, drop pad row.
// ---------------------------------------------------------------------------
__global__ __launch_bounds__(128) void k_outproj(const __bf16* __restrict__ ao,
                                                 const __bf16* __restrict__ wot,
                                                 const float* __restrict__ bias,
                                                 float* __restrict__ out) {
  int wave = threadIdx.x >> 5, lane = threadIdx.x & 31;
  int tile = blockIdx.x * 4 + wave;          // 320 * 8 = 2560 tiles
  int mt = tile % 320, ns = tile / 320;
  int m0 = mt * 16, n0 = ns * 64;
  int lr = lane & 15, lh = lane >> 4;

  floatx8 acc0 = {}, acc1 = {}, acc2 = {}, acc3 = {};
  const __bf16* arow = ao + (size_t)(m0 + lr) * INNER_ + lh * 16;
  const __bf16* b0 = wot + (size_t)(n0 +  0 + lr) * INNER_ + lh * 16;
  const __bf16* b1 = wot + (size_t)(n0 + 16 + lr) * INNER_ + lh * 16;
  const __bf16* b2 = wot + (size_t)(n0 + 32 + lr) * INNER_ + lh * 16;
  const __bf16* b3 = wot + (size_t)(n0 + 48 + lr) * INNER_ + lh * 16;

  for (int k0 = 0; k0 < INNER_; k0 += 32) {
    __builtin_prefetch(arow + k0 + 64, 0, 1);
    bf16x16 a = load_a_frag(arow + k0);
    acc0 = wmma_bf16(a, load_b_frag(b0 + k0), acc0);
    acc1 = wmma_bf16(a, load_b_frag(b1 + k0), acc1);
    acc2 = wmma_bf16(a, load_b_frag(b2 + k0), acc2);
    acc3 = wmma_bf16(a, load_b_frag(b3 + k0), acc3);
  }

  floatx8 accs[4] = {acc0, acc1, acc2, acc3};
#pragma unroll
  for (int t = 0; t < 4; ++t) {
#pragma unroll
    for (int r = 0; r < 8; ++r) {
      int m = m0 + r + 8 * lh;
      int s = m % L_, b = m / L_;
      if (s < N_) {
        int n = n0 + t * 16 + lr;
        out[((size_t)b * N_ + s) * DIM_ + n] = accs[t][r] + bias[n];
      }
    }
  }
}

// ---------------------------------------------------------------------------
// Host-side launcher.
// ---------------------------------------------------------------------------
extern "C" void kernel_launch(void* const* d_in, const int* in_sizes, int n_in,
                              void* d_out, int out_size, void* d_ws, size_t ws_size,
                              hipStream_t stream) {
  (void)in_sizes; (void)n_in; (void)out_size; (void)ws_size;
  const float* x            = (const float*)d_in[0];
  const unsigned char* mask = (const unsigned char*)d_in[1];   // jnp.bool_
  const float* wqkv         = (const float*)d_in[2];
  const float* wout         = (const float*)d_in[3];
  const float* bout         = (const float*)d_in[4];
  float* out = (float*)d_out;

  // Workspace carve-up (all bf16, ~28 MB total, everything 32B-aligned).
  __bf16* p     = (__bf16*)d_ws;
  __bf16* xb    = p; p += (size_t)B_ * L_ * DIM_;      // padded activations
  __bf16* wqkvt = p; p += (size_t)QKV3_ * DIM_;        // W_qkv^T (N-major)
  __bf16* woutt = p; p += (size_t)INNER_ * DIM_;       // W_out^T (N-major)
  __bf16* qh    = p; p += (size_t)BH_ * L_ * DH_;      // q, head-major, *scale
  __bf16* kh    = p; p += (size_t)BH_ * L_ * DH_;      // k, head-major
  __bf16* vt    = p; p += (size_t)BH_ * DH_ * L_;      // v, transposed [d][seq]
  __bf16* ao    = p;                                   // attention out [b*L][INNER]

  k_convert  <<<dim3(4096), dim3(256), 0, stream>>>(x, wqkv, wout, xb, wqkvt, woutt);
  k_qkv      <<<dim3(1920), dim3(128), 0, stream>>>(xb, wqkvt, qh, kh, vt);
  k_attn_text<<<dim3(128),  dim3(128), 0, stream>>>(qh, kh, vt, ao);
  k_attn_img <<<dim3(512),  dim3(128), 0, stream>>>(qh, kh, vt, mask, ao);
  k_outproj  <<<dim3(640),  dim3(128), 0, stream>>>(ao, woutt, bout, out);
}
